// Multi_Head_Attention_56014963474469
// MI455X (gfx1250) — matrix operand order repaired and checked by reading, not compiled, not run on hardware
//
#include <hip/hip_runtime.h>
#include <hip/hip_bf16.h>

// MHA forward, CDNA5 (gfx1250), wave32, f16 WMMA with f32 accumulation.
// GEMMs use explicit ping-pong double buffering; __launch_bounds__(128,1)
// gives each wave the full VGPR budget so accumulators and both fragment
// sets live in fixed registers (no live-range-split copies in the hot loop).
//
// Pipeline (intermediates in d_ws, 52 MB < 192 MB L2):
//   cvt:   f32 -> f16 (vectorized) for inputs_q, inputs_kv, Wp, Wo
//   proj:  Q = Xq @ Wp^T  -> [B,H,S,64]
//          P = Xkv @ Wp^T -> K [B,H,S,64] and Vt [B,H,64,S]  (K == V)
//   attn:  flash attention, one wave per 16-row q tile, causal k bound
//   out:   Out = Xattn @ Wo^T + bo (f32)

typedef __attribute__((ext_vector_type(16))) _Float16 v16h;
typedef __attribute__((ext_vector_type(8)))  _Float16 v8h;
typedef __attribute__((ext_vector_type(4)))  _Float16 v4h;
typedef __attribute__((ext_vector_type(8)))  float    v8f;

#define D_MODEL   1024
#define NUM_HEADS 16
#define HEAD_DIM  64
#define SEQ       2048
#define BATCH     2
#define NEG_BIG   -1.0e10f

union V16U { v16h v; v8h h[2]; };

__device__ __forceinline__ v8f wmma_f16(v16h a, v16h b, v8f c) {
  // (neg_a, A, neg_b, B, c_mod, C, reuse_a, reuse_b)
  return __builtin_amdgcn_wmma_f32_16x16x32_f16(false, a, false, b, (short)0, c,
                                                false, false);
}

// A-matrix 16x32 f16 fragment. `base` = this lane's M-row pointer + k0.
// g = lane>>4. v0..3 hold K = 8g+2v+h ; v4..7 hold K = 16+8g+2(v-4)+h.
template <typename PT>
__device__ __forceinline__ v16h load_a_frag(const PT* base, int g) {
  V16U u;
  u.h[0] = *(const v8h*)(base + 8 * g);
  u.h[1] = *(const v8h*)(base + 16 + 8 * g);
  return u.v;
}

// B-matrix 32x16 f16 fragment. `base` = column-owner row (N = lane&15) + k0.
// g = lane>>4. v0..3 hold K = 8g+2v+h ; v4..7 hold K = 16+8g+2(v-4)+h.
template <typename PT>
__device__ __forceinline__ v16h load_b_frag(const PT* base, int g) {
  V16U u;
  u.h[0] = *(const v8h*)(base + 8 * g);
  u.h[1] = *(const v8h*)(base + 16 + 8 * g);
  return u.v;
}

// Load one full 32x64 x 32k fragment set (2 A-frags + 4 B-frags).
__device__ __forceinline__ void load_set(const _Float16* __restrict__ arow0,
                                         const _Float16* __restrict__ arow1,
                                         const _Float16* __restrict__ wrow,
                                         int k, int g,
                                         v16h& a0, v16h& a1,
                                         v16h& b0, v16h& b1, v16h& b2, v16h& b3) {
  a0 = load_a_frag(arow0 + k, g);
  a1 = load_a_frag(arow1 + k, g);
  b0 = load_b_frag(wrow + 0 * 16 * D_MODEL + k, g);
  b1 = load_b_frag(wrow + 1 * 16 * D_MODEL + k, g);
  b2 = load_b_frag(wrow + 2 * 16 * D_MODEL + k, g);
  b3 = load_b_frag(wrow + 3 * 16 * D_MODEL + k, g);
}

// 8 WMMAs of one 32x64 tile k-step.
__device__ __forceinline__ void mma_set(v8f acc[8],
                                        v16h a0, v16h a1,
                                        v16h b0, v16h b1, v16h b2, v16h b3) {
  acc[0] = wmma_f16(a0, b0, acc[0]);
  acc[1] = wmma_f16(a1, b0, acc[1]);
  acc[2] = wmma_f16(a0, b1, acc[2]);
  acc[3] = wmma_f16(a1, b1, acc[3]);
  acc[4] = wmma_f16(a0, b2, acc[4]);
  acc[5] = wmma_f16(a1, b2, acc[5]);
  acc[6] = wmma_f16(a0, b3, acc[6]);
  acc[7] = wmma_f16(a1, b3, acc[7]);
}

// Full k reduction with ping-pong buffers; no fragment rotation copies.
__device__ __forceinline__ void gemm_k_loop(const _Float16* __restrict__ arow0,
                                            const _Float16* __restrict__ arow1,
                                            const _Float16* __restrict__ wrow,
                                            int g, v8f acc[8]) {
  v16h a0A, a1A, b0A, b1A, b2A, b3A;
  v16h a0B, a1B, b0B, b1B, b2B, b3B;
  load_set(arow0, arow1, wrow, 0, g, a0A, a1A, b0A, b1A, b2A, b3A);
  int k0 = 0;
#pragma unroll 1
  for (; k0 < D_MODEL - 64; k0 += 64) {
    load_set(arow0, arow1, wrow, k0 + 32, g, a0B, a1B, b0B, b1B, b2B, b3B);
    mma_set(acc, a0A, a1A, b0A, b1A, b2A, b3A);
    load_set(arow0, arow1, wrow, k0 + 64, g, a0A, a1A, b0A, b1A, b2A, b3A);
    mma_set(acc, a0B, a1B, b0B, b1B, b2B, b3B);
  }
  // k0 == D_MODEL-64: steps k0 (A set) and k0+32 remain.
  load_set(arow0, arow1, wrow, k0 + 32, g, a0B, a1B, b0B, b1B, b2B, b3B);
  mma_set(acc, a0A, a1A, b0A, b1A, b2A, b3A);
  mma_set(acc, a0B, a1B, b0B, b1B, b2B, b3B);
}

// ---------------------------------------------------------------- convert
__global__ void cvt_f32f16_kernel(const float* __restrict__ src,
                                  _Float16* __restrict__ dst, int n4) {
  int i = blockIdx.x * blockDim.x + threadIdx.x;
  if (i < n4) {
    float4 f = ((const float4*)src)[i];
    v4h h;
    h.x = (_Float16)f.x; h.y = (_Float16)f.y;
    h.z = (_Float16)f.z; h.w = (_Float16)f.w;
    ((v4h*)dst)[i] = h;
  }
}

// ---------------------------------------------------------------- projection
// Wave tile: 32 tokens x 64 outputs. C[t,o] = sum_i A[t,i] * W[o,i].
// outK:  [B,H,S,64]  — always written.  outVt: [B,H,64,S] — when non-null.
__global__ __launch_bounds__(128, 1)
void proj_gemm_kernel(const _Float16* __restrict__ A,
                      const _Float16* __restrict__ W,
                      _Float16* __restrict__ outK,
                      _Float16* __restrict__ outVt) {
  const int lane = threadIdx.x & 31;
  const int wid  = threadIdx.x >> 5;
  const int n16  = lane & 15;
  const int g    = lane >> 4;

  const int wgl   = blockIdx.x * 4 + wid;   // 0..2047
  const int mtile = wgl >> 4;               // 128 token tiles of 32
  const int ntile = wgl & 15;               // 16 output tiles of 64
  const int obase = ntile * 64;

  const _Float16* arow0 = A + (size_t)(mtile * 32 + n16) * D_MODEL;
  const _Float16* arow1 = arow0 + 16 * D_MODEL;
  const _Float16* wrow  = W + (size_t)(obase + n16) * D_MODEL;

  v8f acc[8] = {v8f{}, v8f{}, v8f{}, v8f{}, v8f{}, v8f{}, v8f{}, v8f{}};
  gemm_k_loop(arow0, arow1, wrow, g, acc);

#pragma unroll
  for (int j = 0; j < 4; ++j) {
    const int o = obase + 16 * j + n16;
    const int h = o >> 6;
    const int d = o & 63;
#pragma unroll
    for (int i = 0; i < 2; ++i) {
      const v8f& cacc = acc[(j << 1) | i];
      const int tbase = mtile * 32 + i * 16 + 8 * g;   // rows tbase..tbase+7
      const int b = tbase >> 11;                       // SEQ = 2048
      const int s = tbase & (SEQ - 1);
      const size_t bh = (size_t)(b * NUM_HEADS + h);
#pragma unroll
      for (int r = 0; r < 8; ++r)
        outK[(bh * SEQ + (s + r)) * HEAD_DIM + d] = (_Float16)cacc[r];
      if (outVt) {
        v8h p;
#pragma unroll
        for (int r = 0; r < 8; ++r) p[r] = (_Float16)cacc[r];
        *(v8h*)&outVt[(bh * HEAD_DIM + d) * SEQ + s] = p;   // s contiguous
      }
    }
  }
}

// ---------------------------------------------------------------- output GEMM
__global__ __launch_bounds__(128, 1)
void out_gemm_kernel(const _Float16* __restrict__ A,
                     const _Float16* __restrict__ W,
                     const float* __restrict__ bias,
                     float* __restrict__ out) {
  const int lane = threadIdx.x & 31;
  const int wid  = threadIdx.x >> 5;
  const int n16  = lane & 15;
  const int g    = lane >> 4;

  const int wgl   = blockIdx.x * 4 + wid;
  const int mtile = wgl >> 4;
  const int ntile = wgl & 15;
  const int obase = ntile * 64;

  const _Float16* arow0 = A + (size_t)(mtile * 32 + n16) * D_MODEL;
  const _Float16* arow1 = arow0 + 16 * D_MODEL;
  const _Float16* wrow  = W + (size_t)(obase + n16) * D_MODEL;

  v8f acc[8] = {v8f{}, v8f{}, v8f{}, v8f{}, v8f{}, v8f{}, v8f{}, v8f{}};
  gemm_k_loop(arow0, arow1, wrow, g, acc);

#pragma unroll
  for (int j = 0; j < 4; ++j) {
    const int o = obase + 16 * j + n16;
    const float bval = bias[o];
#pragma unroll
    for (int i = 0; i < 2; ++i) {
      const v8f& cacc = acc[(j << 1) | i];
      const int tbase = mtile * 32 + i * 16 + 8 * g;
#pragma unroll
      for (int r = 0; r < 8; ++r)
        out[(size_t)(tbase + r) * D_MODEL + o] = cacc[r] + bval;
    }
  }
}

// ---------------------------------------------------------------- flash attention
// One wave = one 16-row q tile of one (b,h). 32-key blocks, causal bound.
// Next-step K fragments and current V fragments are issued before the softmax
// VALU section so their latency hides under exp/shuffle work.
__global__ __launch_bounds__(128, 1)
void attn_fa_kernel(const _Float16* __restrict__ Q,
                    const _Float16* __restrict__ K,
                    const _Float16* __restrict__ Vt,
                    _Float16* __restrict__ X) {
  const int lane = threadIdx.x & 31;
  const int wid  = threadIdx.x >> 5;
  const int n16  = lane & 15;
  const int g    = lane >> 4;

  const int qblocks = SEQ / 64;                     // 32
  const int bh    = blockIdx.x / qblocks;           // b*16 + h
  const int qb    = blockIdx.x % qblocks;
  const int qbase = qb * 64 + wid * 16;

  const _Float16* Qh = Q  + (size_t)bh * SEQ * HEAD_DIM;
  const _Float16* Kh = K  + (size_t)bh * SEQ * HEAD_DIM;
  const _Float16* Vh = Vt + (size_t)bh * HEAD_DIM * SEQ;

  // Per-wave LDS tile for P (16 rows x 32 keys, row stride 40 halves so
  // ds_load_b128 reads stay 16B-aligned).
  __shared__ __align__(16) _Float16 lds[4][16 * 40];
  _Float16* P = lds[wid];

  const _Float16* qrow = Qh + (size_t)(qbase + n16) * HEAD_DIM;
  const v16h aq0 = load_a_frag(qrow, g);
  const v16h aq1 = load_a_frag(qrow + 32, g);

  v8f acc[4] = {v8f{}, v8f{}, v8f{}, v8f{}};       // x tile: 16 q x 64 d
  float mrun[8], lrun[8];
#pragma unroll
  for (int r = 0; r < 8; ++r) { mrun[r] = -3.0e38f; lrun[r] = 0.0f; }

  // Prologue: K fragments for kbase = 0.
  const _Float16* kr0 = Kh + (size_t)n16 * HEAD_DIM;
  v16h kf0 = load_b_frag(kr0, g);
  v16h kf1 = load_b_frag(kr0 + 32, g);
  v16h kf2 = load_b_frag(kr0 + 16 * HEAD_DIM, g);
  v16h kf3 = load_b_frag(kr0 + 16 * HEAD_DIM + 32, g);

  const int nsteps = (qbase + 16 + 31) >> 5;       // causal: keys 0..qbase+15
#pragma unroll 1
  for (int kb = 0; kb < nsteps; ++kb) {
    const int kbase = kb * 32;

    // ---- scores with already-resident K fragments
    v8f s0 = {}, s1 = {};
    s0 = wmma_f16(aq0, kf0, s0);
    s0 = wmma_f16(aq1, kf1, s0);
    s1 = wmma_f16(aq0, kf2, s1);
    s1 = wmma_f16(aq1, kf3, s1);

    // ---- issue current V loads + next-step K loads (hidden under softmax)
    v16h vf0 = load_b_frag(Vh + (size_t)(0  + n16) * SEQ + kbase, g);
    v16h vf1 = load_b_frag(Vh + (size_t)(16 + n16) * SEQ + kbase, g);
    v16h vf2 = load_b_frag(Vh + (size_t)(32 + n16) * SEQ + kbase, g);
    v16h vf3 = load_b_frag(Vh + (size_t)(48 + n16) * SEQ + kbase, g);
    {
      const int kpre = (kbase + 32 < SEQ) ? (kbase + 32) : (SEQ - 32);  // branchless, in-bounds
      const _Float16* krn = Kh + (size_t)(kpre + n16) * HEAD_DIM;
      kf0 = load_b_frag(krn, g);
      kf1 = load_b_frag(krn + 32, g);
      kf2 = load_b_frag(krn + 16 * HEAD_DIM, g);
      kf3 = load_b_frag(krn + 16 * HEAD_DIM + 32, g);
    }

    // ---- causal bias + online softmax (rows m = r + 8g, cols = lane&15)
#pragma unroll
    for (int r = 0; r < 8; ++r) {
      const int qrow_i = qbase + r + 8 * g;
      float a0 = (kbase + n16)      <= qrow_i ? (float)s0[r] : NEG_BIG;
      float a1 = (kbase + 16 + n16) <= qrow_i ? (float)s1[r] : NEG_BIG;

      float mt = fmaxf(a0, a1);
      mt = fmaxf(mt, __shfl_xor(mt, 1, 32));
      mt = fmaxf(mt, __shfl_xor(mt, 2, 32));
      mt = fmaxf(mt, __shfl_xor(mt, 4, 32));
      mt = fmaxf(mt, __shfl_xor(mt, 8, 32));

      float mn = fmaxf(mrun[r], mt);
      float sc = __expf(mrun[r] - mn);
      mrun[r] = mn;

      float p0 = __expf(a0 - mn);
      float p1 = __expf(a1 - mn);
      float rs = p0 + p1;
      rs += __shfl_xor(rs, 1, 32);
      rs += __shfl_xor(rs, 2, 32);
      rs += __shfl_xor(rs, 4, 32);
      rs += __shfl_xor(rs, 8, 32);
      lrun[r] = lrun[r] * sc + rs;

      acc[0][r] *= sc; acc[1][r] *= sc; acc[2][r] *= sc; acc[3][r] *= sc;

      // C-layout -> row-major LDS (transpose staging for the P A-fragment)
      const int row = r + 8 * g;
      P[row * 40 + n16]      = (_Float16)p0;
      P[row * 40 + 16 + n16] = (_Float16)p1;
    }
    // Same-wave LDS ops complete in order; just stop compiler reordering.
    __builtin_amdgcn_wave_barrier();

    // ---- x += P(16x32) @ V(32x64)
    const v16h pa = load_a_frag(P + n16 * 40, g);
    acc[0] = wmma_f16(pa, vf0, acc[0]);
    acc[1] = wmma_f16(pa, vf1, acc[1]);
    acc[2] = wmma_f16(pa, vf2, acc[2]);
    acc[3] = wmma_f16(pa, vf3, acc[3]);
  }

  // ---- normalize and write X[b, q, h*64 + d] (f16)
  float inv[8];
#pragma unroll
  for (int r = 0; r < 8; ++r) inv[r] = 1.0f / lrun[r];

  const int b = bh >> 4;
  const int h = bh & 15;
  _Float16* Xp = X + ((size_t)b * SEQ + qbase) * D_MODEL + h * HEAD_DIM;
#pragma unroll
  for (int c = 0; c < 4; ++c) {
#pragma unroll
    for (int r = 0; r < 8; ++r) {
      const int row = r + 8 * g;
      const int d = 16 * c + n16;
      Xp[(size_t)row * D_MODEL + d] = (_Float16)(acc[c][r] * inv[r]);
    }
  }
}

// ---------------------------------------------------------------- host
extern "C" void kernel_launch(void* const* d_in, const int* in_sizes, int n_in,
                              void* d_out, int out_size, void* d_ws, size_t ws_size,
                              hipStream_t stream) {
  const float* inputs_q  = (const float*)d_in[0];   // [2,2048,1024]
  const float* inputs_kv = (const float*)d_in[1];   // [2,2048,1024]
  // d_in[2] = mask: causal tril, reproduced analytically in attn_fa_kernel
  const float* Wp = (const float*)d_in[3];          // [1024,1024]
  const float* Wo = (const float*)d_in[4];          // [1024,1024]
  const float* bo = (const float*)d_in[5];          // [1024]
  float* out = (float*)d_out;

  char* ws = (char*)d_ws;
  const size_t MB = (size_t)1 << 20;
  _Float16* Wp16 = (_Float16*)(ws + 0 * MB);   //  2 MB
  _Float16* Wo16 = (_Float16*)(ws + 2 * MB);   //  2 MB
  _Float16* XQ   = (_Float16*)(ws + 4 * MB);   //  8 MB
  _Float16* XKV  = (_Float16*)(ws + 12 * MB);  //  8 MB
  _Float16* Qb   = (_Float16*)(ws + 20 * MB);  //  8 MB [B,H,S,64]
  _Float16* Kb   = (_Float16*)(ws + 28 * MB);  //  8 MB [B,H,S,64]
  _Float16* VTb  = (_Float16*)(ws + 36 * MB);  //  8 MB [B,H,64,S]
  _Float16* XA   = (_Float16*)(ws + 44 * MB);  //  8 MB [B,S,1024]

  const int nW4 = D_MODEL * D_MODEL / 4;       // 256K float4
  const int nX4 = BATCH * SEQ * D_MODEL / 4;   // 1M float4
  cvt_f32f16_kernel<<<(nW4 + 255) / 256, 256, 0, stream>>>(Wp, Wp16, nW4);
  cvt_f32f16_kernel<<<(nW4 + 255) / 256, 256, 0, stream>>>(Wo, Wo16, nW4);
  cvt_f32f16_kernel<<<(nX4 + 255) / 256, 256, 0, stream>>>(inputs_q, XQ, nX4);
  cvt_f32f16_kernel<<<(nX4 + 255) / 256, 256, 0, stream>>>(inputs_kv, XKV, nX4);

  // 4096 tokens / 32 * (1024/64) = 2048 wave tiles -> 512 blocks of 4 waves
  proj_gemm_kernel<<<512, 128, 0, stream>>>(XQ, Wp16, Qb, (_Float16*)nullptr);
  proj_gemm_kernel<<<512, 128, 0, stream>>>(XKV, Wp16, Kb, VTb);

  // B*H*(SEQ/64) = 2*16*32 = 1024 blocks, 4 waves each own a 16-row q tile
  attn_fa_kernel<<<1024, 128, 0, stream>>>(Qb, Kb, VTb, XA);

  out_gemm_kernel<<<512, 128, 0, stream>>>(XA, Wo16, bo, out);
}
